// BidirectionalLSTM_skip_45930380264311
// MI455X (gfx1250) — compile-verified, matrix operand order checked by reading, not run on hardware
//
#include <hip/hip_runtime.h>

// Problem dims (fixed by reference)
#define B_SZ 256
#define T_SZ 26
#define H_SZ 512   // I == H == O == 512
#define NBLK_REC 64   // blocks in persistent recurrent kernel (8 waves each)

typedef __attribute__((ext_vector_type(16))) __bf16 v16bf;
typedef __attribute__((ext_vector_type(8)))  float  v8f;

// ---------------------------------------------------------------------------
// WMMA 16-bit operand fragment layout (wave32), per ISA:
//   lane L (L<16): row = L,     elems 0..7 = K+0..7,   elems 8..15 = K+16..23
//   lane L (>=16): row = L-16,  elems 0..7 = K+8..15,  elems 8..15 = K+24..31
// koff = (lane>=16) ? 8 : 0.
// ---------------------------------------------------------------------------

struct Raw8 { float4 a0, a1, b0, b1; };

__device__ __forceinline__ Raw8 load_raw(const float* __restrict__ rowptr, int koff) {
    Raw8 r;
    const float4* pa = reinterpret_cast<const float4*>(rowptr + koff);
    const float4* pb = reinterpret_cast<const float4*>(rowptr + koff + 16);
    r.a0 = pa[0]; r.a1 = pa[1]; r.b0 = pb[0]; r.b1 = pb[1];
    return r;
}

__device__ __forceinline__ v16bf cvt_raw(const Raw8& r) {
    v16bf v;
    v[0]  = (__bf16)r.a0.x; v[1]  = (__bf16)r.a0.y; v[2]  = (__bf16)r.a0.z; v[3]  = (__bf16)r.a0.w;
    v[4]  = (__bf16)r.a1.x; v[5]  = (__bf16)r.a1.y; v[6]  = (__bf16)r.a1.z; v[7]  = (__bf16)r.a1.w;
    v[8]  = (__bf16)r.b0.x; v[9]  = (__bf16)r.b0.y; v[10] = (__bf16)r.b0.z; v[11] = (__bf16)r.b0.w;
    v[12] = (__bf16)r.b1.x; v[13] = (__bf16)r.b1.y; v[14] = (__bf16)r.b1.z; v[15] = (__bf16)r.b1.w;
    return v;
}

__device__ __forceinline__ v16bf cvt_frag(const float* __restrict__ rowptr, int koff) {
    return cvt_raw(load_raw(rowptr, koff));
}

// Fragment load from a pre-swizzled bf16 buffer:
// addr = ((r0*ktiles + k0)*32 + lane) * 16 elements.
__device__ __forceinline__ v16bf load_sw(const __bf16* __restrict__ base,
                                         int ktiles, int r0, int k0, int lane) {
    return *reinterpret_cast<const v16bf*>(
        base + ((((size_t)r0 * ktiles + k0) * 32 + lane) << 4));
}

__device__ __forceinline__ v8f wmma_bf16(v16bf a, v16bf b, v8f c) {
    return __builtin_amdgcn_wmma_f32_16x16x32_bf16(false, a, false, b, (short)0, c, false, false);
}

__device__ __forceinline__ float fsig(float x) {
    return 1.0f / (1.0f + __expf(-x));
}
__device__ __forceinline__ float ftanh(float x) {
    float ax = fabsf(x);
    float e  = __expf(-2.0f * ax);
    float t  = (1.0f - e) / (1.0f + e);
    return copysignf(t, x);
}

// ---------------------------------------------------------------------------
// Pre-convert fp32 row-major [rows, kdim] -> bf16 fragment-swizzled buffer.
// ---------------------------------------------------------------------------
__global__ __launch_bounds__(256) void swizzle_bf16_kernel(
    const float* __restrict__ src, __bf16* __restrict__ dst,
    int kdim, long long total) {
    long long tid = (long long)blockIdx.x * 256 + threadIdx.x;
    if (tid >= total) return;
    const int  lane   = (int)(tid & 31);
    long long  fl     = tid >> 5;
    const int  ktiles = kdim >> 5;
    const int  k0     = (int)(fl % ktiles);
    long long  r0     = fl / ktiles;
    const float* rowptr = src + (size_t)(r0 * 16 + (lane & 15)) * kdim + (size_t)k0 * 32;
    v16bf f = cvt_frag(rowptr, (lane >> 4) << 3);
    *reinterpret_cast<v16bf*>(dst + (tid << 4)) = f;
}

__global__ __launch_bounds__(64) void zero_bar_kernel(unsigned* __restrict__ bar) {
    bar[threadIdx.x] = 0u;
}

// ---------------------------------------------------------------------------
// Parallel pre-GEMM: gates_pre[t] = x_t @ W_ih^T + b_ih + b_hh   (all t).
// Wave tile: 16 rows x 64 gate-cols. Grid: (64, T_SZ), 256 threads.
// ---------------------------------------------------------------------------
__global__ __launch_bounds__(256) void gates_pre_kernel(
    const __bf16* __restrict__ xb,    // swizzled [256 x T*512]
    const __bf16* __restrict__ Wihb,  // swizzled [T*2048 x 512]
    const float*  __restrict__ b_ih,  // [T,4H]
    const float*  __restrict__ b_hh,  // [T,4H]
    float* __restrict__ gp) {         // [T, B, 4H]
    const int lane = threadIdx.x & 31;
    const int wid  = threadIdx.x >> 5;
    const int t    = blockIdx.y;
    const int gw   = blockIdx.x * 8 + wid;   // 0..511
    const int m0   = (gw >> 5) << 4;         // row tile
    const int n0w  = (gw & 31) << 6;         // 64-col base in [0,2048)
    const int lr   = lane & 15;

    v8f zero = {};
    v8f acc[4];
    #pragma unroll
    for (int g = 0; g < 4; ++g) acc[g] = zero;

    const int ktA = T_SZ * 16;
    const int r0a = m0 >> 4;
    const int kb  = t * 16;
    const int rw0 = (t * 4 * H_SZ + n0w) >> 4;       // 4 adjacent 16-row weight tiles
    const int rw1 = rw0 + 1, rw2 = rw0 + 2, rw3 = rw0 + 3;

    v16bf a  = load_sw(xb,   ktA, r0a, kb, lane);
    v16bf b0 = load_sw(Wihb, 16,  rw0, 0, lane);
    v16bf b1 = load_sw(Wihb, 16,  rw1, 0, lane);
    v16bf b2 = load_sw(Wihb, 16,  rw2, 0, lane);
    v16bf b3 = load_sw(Wihb, 16,  rw3, 0, lane);
    for (int kk = 0; kk < 16; ++kk) {
        const int kn = (kk + 1) & 15;
        v16bf an = load_sw(xb,   ktA, r0a, kb + kn, lane);
        v16bf c0 = load_sw(Wihb, 16,  rw0, kn, lane);
        v16bf c1 = load_sw(Wihb, 16,  rw1, kn, lane);
        v16bf c2 = load_sw(Wihb, 16,  rw2, kn, lane);
        v16bf c3 = load_sw(Wihb, 16,  rw3, kn, lane);
        acc[0] = wmma_bf16(a, b0, acc[0]);
        acc[1] = wmma_bf16(a, b1, acc[1]);
        acc[2] = wmma_bf16(a, b2, acc[2]);
        acc[3] = wmma_bf16(a, b3, acc[3]);
        a = an; b0 = c0; b1 = c1; b2 = c2; b3 = c3;
    }

    const int rbase = m0 + ((lane >> 4) << 3);
    #pragma unroll
    for (int g = 0; g < 4; ++g) {
        const int col = n0w + g * 16 + lr;                 // 0..2047
        const float bb = b_ih[t * 4 * H_SZ + col] + b_hh[t * 4 * H_SZ + col];
        #pragma unroll
        for (int r = 0; r < 8; ++r) {
            gp[((size_t)t * B_SZ + rbase + r) * (4 * H_SZ) + col] = acc[g][r] + bb;
        }
    }
}

// ---------------------------------------------------------------------------
// Device-scope grid barrier (blocks are co-resident: 64 blocks x 8 waves).
// ---------------------------------------------------------------------------
__device__ __forceinline__ void grid_barrier(unsigned* __restrict__ bar, int t) {
    __threadfence();   // release this wave's stores to device scope
    __syncthreads();
    if (threadIdx.x == 0) {
        __hip_atomic_fetch_add(&bar[t], 1u, __ATOMIC_RELEASE, __HIP_MEMORY_SCOPE_AGENT);
        while (__hip_atomic_load(&bar[t], __ATOMIC_ACQUIRE, __HIP_MEMORY_SCOPE_AGENT)
               < (unsigned)NBLK_REC) {
            __builtin_amdgcn_s_sleep(1);
        }
    }
    __syncthreads();
    __threadfence();   // acquire: invalidate stale lines before reading new h
}

// ---------------------------------------------------------------------------
// Persistent recurrent kernel: all T steps in one launch.
// Per step: gates = gates_pre + h @ W_hh^T, activations, skip, state update.
// Grid: NBLK_REC blocks x 256 threads = 512 waves (one 16x16 tile each).
// ---------------------------------------------------------------------------
__global__ __launch_bounds__(256) void lstm_recurrent_kernel(
    const __bf16* __restrict__ Whhb,  // swizzled [T*2048 x 512]
    const float*  __restrict__ gp,    // [T, B, 4H] (biases included)
    float* __restrict__ hbuf,         // 2 x [B,H]
    float* __restrict__ cbuf,         // 2 x [B,H]
    float* __restrict__ hs,           // [B, T, H]
    unsigned* __restrict__ bar) {
    const int lane = threadIdx.x & 31;
    const int wid  = threadIdx.x >> 5;
    const int gw   = blockIdx.x * 8 + wid;   // 0..511
    const int m0   = (gw >> 5) << 4;
    const int n0   = (gw & 31) << 4;
    const int lr   = lane & 15;
    const int koff = (lane >> 4) << 3;
    const int col  = n0 + lr;
    const int rbase = m0 + ((lane >> 4) << 3);
    const size_t bh = (size_t)B_SZ * H_SZ;

    for (int t = 0; t < T_SZ; ++t) {
        const int first = (t == 0);
        const float* h_in  = hbuf + ((t + 1) & 1) * bh;
        const float* c_in  = cbuf + ((t + 1) & 1) * bh;
        float*       h_out = hbuf + (t & 1) * bh;
        float*       c_out = cbuf + (t & 1) * bh;

        // Prefetch gates_pre for this tile (latency hides under the GEMM).
        float gpv[4][8];
        #pragma unroll
        for (int g = 0; g < 4; ++g) {
            #pragma unroll
            for (int r = 0; r < 8; ++r) {
                gpv[g][r] = gp[((size_t)t * B_SZ + rbase + r) * (4 * H_SZ) + g * H_SZ + col];
            }
        }

        v8f zero = {};
        v8f acc[4];
        #pragma unroll
        for (int g = 0; g < 4; ++g) acc[g] = zero;

        if (!first) {  // wave-uniform; EXEC all-ones around WMMA
            const int rw0 = (t * 4 * H_SZ + n0) >> 4;
            const int rw1 = rw0 + 32, rw2 = rw0 + 64, rw3 = rw0 + 96;  // +512 rows each
            const float* hrow = h_in + (size_t)(m0 + lr) * H_SZ;
            v16bf a  = cvt_frag(hrow, koff);
            v16bf b0 = load_sw(Whhb, 16, rw0, 0, lane);
            v16bf b1 = load_sw(Whhb, 16, rw1, 0, lane);
            v16bf b2 = load_sw(Whhb, 16, rw2, 0, lane);
            v16bf b3 = load_sw(Whhb, 16, rw3, 0, lane);
            for (int kk = 0; kk < 16; ++kk) {
                const int kn = (kk + 1) & 15;
                Raw8  ra = load_raw(hrow + kn * 32, koff);
                v16bf c0 = load_sw(Whhb, 16, rw0, kn, lane);
                v16bf c1 = load_sw(Whhb, 16, rw1, kn, lane);
                v16bf c2 = load_sw(Whhb, 16, rw2, kn, lane);
                v16bf c3 = load_sw(Whhb, 16, rw3, kn, lane);
                acc[0] = wmma_bf16(a, b0, acc[0]);
                acc[1] = wmma_bf16(a, b1, acc[1]);
                acc[2] = wmma_bf16(a, b2, acc[2]);
                acc[3] = wmma_bf16(a, b3, acc[3]);
                a = cvt_raw(ra);
                b0 = c0; b1 = c1; b2 = c2; b3 = c3;
            }
        }

        #pragma unroll
        for (int r = 0; r < 8; ++r) {
            const int row = rbase + r;
            float iv = fsig(acc[0][r] + gpv[0][r]);
            float fv = fsig(acc[1][r] + gpv[1][r]);
            float gv = ftanh(acc[2][r] + gpv[2][r]);
            float ov = fsig(acc[3][r] + gpv[3][r]);
            float cp = first ? 0.0f : c_in[(size_t)row * H_SZ + col];
            float hp = first ? 0.0f : h_in[(size_t)row * H_SZ + col];
            float cn = fv * cp + iv * gv;
            float hn = ov * ftanh(cn);
            hn += first ? 0.0f : hp;
            cn += first ? 0.0f : cp;
            h_out[(size_t)row * H_SZ + col] = hn;
            c_out[(size_t)row * H_SZ + col] = cn;
            hs[(size_t)row * (T_SZ * H_SZ) + (size_t)t * H_SZ + col] = hn;
        }

        if (t < T_SZ - 1) grid_barrier(bar, t);
    }
}

// ---------------------------------------------------------------------------
// Per-step kernel (tiers B/C): fused x-GEMM + h-GEMM, one launch per t.
// ---------------------------------------------------------------------------
template <bool PRE>
__global__ __launch_bounds__(256) void lstm_step_kernel(
    const float*  __restrict__ x, const __bf16* __restrict__ xb,
    const float*  __restrict__ W_ih, const __bf16* __restrict__ Wihb,
    const float*  __restrict__ W_hh, const __bf16* __restrict__ Whhb,
    const float*  __restrict__ b_ih, const float* __restrict__ b_hh,
    const float*  __restrict__ h_in, const float* __restrict__ c_in,
    float* __restrict__ h_out, float* __restrict__ c_out,
    float* __restrict__ hs, int t, int first) {
    const int lane = threadIdx.x & 31;
    const int wid  = threadIdx.x >> 5;
    const int gw   = blockIdx.x * 8 + wid;
    const int m0   = (gw >> 5) << 4;
    const int n0   = (gw & 31) << 4;
    const int lr   = lane & 15;
    const int koff = (lane >> 4) << 3;

    v8f zero = {};
    v8f acc[4];
    #pragma unroll
    for (int g = 0; g < 4; ++g) acc[g] = zero;

    const int wbase = t * 4 * H_SZ + n0;
    const int rw0 = wbase >> 4, rw1 = rw0 + 32, rw2 = rw0 + 64, rw3 = rw0 + 96;

    if (PRE) {
        {
            const int ktA = T_SZ * 16, r0a = m0 >> 4, kb = t * 16;
            v16bf a  = load_sw(xb,   ktA, r0a, kb, lane);
            v16bf b0 = load_sw(Wihb, 16,  rw0, 0, lane);
            v16bf b1 = load_sw(Wihb, 16,  rw1, 0, lane);
            v16bf b2 = load_sw(Wihb, 16,  rw2, 0, lane);
            v16bf b3 = load_sw(Wihb, 16,  rw3, 0, lane);
            for (int kk = 0; kk < 16; ++kk) {
                const int kn = (kk + 1) & 15;
                v16bf an = load_sw(xb,   ktA, r0a, kb + kn, lane);
                v16bf c0 = load_sw(Wihb, 16,  rw0, kn, lane);
                v16bf c1 = load_sw(Wihb, 16,  rw1, kn, lane);
                v16bf c2 = load_sw(Wihb, 16,  rw2, kn, lane);
                v16bf c3 = load_sw(Wihb, 16,  rw3, kn, lane);
                acc[0] = wmma_bf16(a, b0, acc[0]);
                acc[1] = wmma_bf16(a, b1, acc[1]);
                acc[2] = wmma_bf16(a, b2, acc[2]);
                acc[3] = wmma_bf16(a, b3, acc[3]);
                a = an; b0 = c0; b1 = c1; b2 = c2; b3 = c3;
            }
        }
        if (!first) {
            const float* hrow = h_in + (size_t)(m0 + lr) * H_SZ;
            v16bf a  = cvt_frag(hrow, koff);
            v16bf b0 = load_sw(Whhb, 16, rw0, 0, lane);
            v16bf b1 = load_sw(Whhb, 16, rw1, 0, lane);
            v16bf b2 = load_sw(Whhb, 16, rw2, 0, lane);
            v16bf b3 = load_sw(Whhb, 16, rw3, 0, lane);
            for (int kk = 0; kk < 16; ++kk) {
                const int kn = (kk + 1) & 15;
                Raw8  ra = load_raw(hrow + kn * 32, koff);
                v16bf c0 = load_sw(Whhb, 16, rw0, kn, lane);
                v16bf c1 = load_sw(Whhb, 16, rw1, kn, lane);
                v16bf c2 = load_sw(Whhb, 16, rw2, kn, lane);
                v16bf c3 = load_sw(Whhb, 16, rw3, kn, lane);
                acc[0] = wmma_bf16(a, b0, acc[0]);
                acc[1] = wmma_bf16(a, b1, acc[1]);
                acc[2] = wmma_bf16(a, b2, acc[2]);
                acc[3] = wmma_bf16(a, b3, acc[3]);
                a = cvt_raw(ra);
                b0 = c0; b1 = c1; b2 = c2; b3 = c3;
            }
        }
    } else {
        const float* arow = x + (size_t)(m0 + lr) * (T_SZ * H_SZ) + (size_t)t * H_SZ;
        for (int kk = 0; kk < 16; ++kk) {
            v16bf a = cvt_frag(arow + kk * 32, koff);
            #pragma unroll
            for (int g = 0; g < 4; ++g) {
                v16bf b = cvt_frag(W_ih + (size_t)(wbase + g * H_SZ + lr) * H_SZ + kk * 32, koff);
                acc[g] = wmma_bf16(a, b, acc[g]);
            }
        }
        if (!first) {
            const float* hrow = h_in + (size_t)(m0 + lr) * H_SZ;
            for (int kk = 0; kk < 16; ++kk) {
                v16bf a = cvt_frag(hrow + kk * 32, koff);
                #pragma unroll
                for (int g = 0; g < 4; ++g) {
                    v16bf b = cvt_frag(W_hh + (size_t)(wbase + g * H_SZ + lr) * H_SZ + kk * 32, koff);
                    acc[g] = wmma_bf16(a, b, acc[g]);
                }
            }
        }
    }

    const int col = n0 + lr;
    const float bi = b_ih[col]            + b_hh[col];
    const float bf = b_ih[H_SZ + col]     + b_hh[H_SZ + col];
    const float bg = b_ih[2 * H_SZ + col] + b_hh[2 * H_SZ + col];
    const float bo = b_ih[3 * H_SZ + col] + b_hh[3 * H_SZ + col];
    const int rbase = m0 + ((lane >> 4) << 3);

    #pragma unroll
    for (int r = 0; r < 8; ++r) {
        const int row = rbase + r;
        float iv = fsig(acc[0][r] + bi);
        float fv = fsig(acc[1][r] + bf);
        float gv = ftanh(acc[2][r] + bg);
        float ov = fsig(acc[3][r] + bo);
        float cp = first ? 0.0f : c_in[(size_t)row * H_SZ + col];
        float hp = first ? 0.0f : h_in[(size_t)row * H_SZ + col];
        float cn = fv * cp + iv * gv;
        float hn = ov * ftanh(cn);
        hn += first ? 0.0f : hp;
        cn += first ? 0.0f : cp;
        h_out[(size_t)row * H_SZ + col] = hn;
        c_out[(size_t)row * H_SZ + col] = cn;
        hs[(size_t)row * (T_SZ * H_SZ) + (size_t)t * H_SZ + col] = hn;
    }
}

// ---------------------------------------------------------------------------
// Exit projection: out[B*T, O] = hs[B*T, H] @ W_exit^T + b_exit.
// ---------------------------------------------------------------------------
template <bool PRE>
__global__ __launch_bounds__(256) void exit_gemm_kernel(
    const float*  __restrict__ hs, const float* __restrict__ W_exit,
    const __bf16* __restrict__ Wexitb, const float* __restrict__ b_exit,
    float* __restrict__ out) {
    const int lane = threadIdx.x & 31;
    const int wid  = threadIdx.x >> 5;
    const int m0   = blockIdx.x << 4;
    const int n0   = wid << 6;
    const int lr   = lane & 15;
    const int koff = (lane >> 4) << 3;

    v8f zero = {};
    v8f acc[4];
    #pragma unroll
    for (int g = 0; g < 4; ++g) acc[g] = zero;

    const float* arow = hs + (size_t)(m0 + lr) * H_SZ;
    const int rw0 = n0 >> 4, rw1 = rw0 + 1, rw2 = rw0 + 2, rw3 = rw0 + 3;

    if (PRE) {
        v16bf a  = cvt_frag(arow, koff);
        v16bf b0 = load_sw(Wexitb, 16, rw0, 0, lane);
        v16bf b1 = load_sw(Wexitb, 16, rw1, 0, lane);
        v16bf b2 = load_sw(Wexitb, 16, rw2, 0, lane);
        v16bf b3 = load_sw(Wexitb, 16, rw3, 0, lane);
        for (int kk = 0; kk < 16; ++kk) {
            const int kn = (kk + 1) & 15;
            Raw8  ra = load_raw(arow + kn * 32, koff);
            v16bf c0 = load_sw(Wexitb, 16, rw0, kn, lane);
            v16bf c1 = load_sw(Wexitb, 16, rw1, kn, lane);
            v16bf c2 = load_sw(Wexitb, 16, rw2, kn, lane);
            v16bf c3 = load_sw(Wexitb, 16, rw3, kn, lane);
            acc[0] = wmma_bf16(a, b0, acc[0]);
            acc[1] = wmma_bf16(a, b1, acc[1]);
            acc[2] = wmma_bf16(a, b2, acc[2]);
            acc[3] = wmma_bf16(a, b3, acc[3]);
            a = cvt_raw(ra);
            b0 = c0; b1 = c1; b2 = c2; b3 = c3;
        }
    } else {
        for (int kk = 0; kk < 16; ++kk) {
            v16bf a = cvt_frag(arow + kk * 32, koff);
            #pragma unroll
            for (int g = 0; g < 4; ++g) {
                v16bf b = cvt_frag(W_exit + (size_t)(n0 + g * 16 + lr) * H_SZ + kk * 32, koff);
                acc[g] = wmma_bf16(a, b, acc[g]);
            }
        }
    }

    const int rbase = m0 + ((lane >> 4) << 3);
    #pragma unroll
    for (int g = 0; g < 4; ++g) {
        const int col  = n0 + g * 16 + lr;
        const float bb = b_exit[col];
        #pragma unroll
        for (int r = 0; r < 8; ++r) {
            out[(size_t)(rbase + r) * H_SZ + col] = acc[g][r] + bb;
        }
    }
}

// ---------------------------------------------------------------------------
extern "C" void kernel_launch(void* const* d_in, const int* in_sizes, int n_in,
                              void* d_out, int out_size, void* d_ws, size_t ws_size,
                              hipStream_t stream) {
    (void)in_sizes; (void)n_in; (void)out_size;

    const float* x      = (const float*)d_in[0];
    const float* W_ih   = (const float*)d_in[1];
    const float* W_hh   = (const float*)d_in[2];
    const float* b_ih   = (const float*)d_in[3];
    const float* b_hh   = (const float*)d_in[4];
    const float* W_exit = (const float*)d_in[5];
    const float* b_exit = (const float*)d_in[6];
    float* out = (float*)d_out;

    // ---- fp32 state region ----
    float* ws   = (float*)d_ws;
    float* hs   = ws;
    float* hbuf = hs + (size_t)B_SZ * T_SZ * H_SZ;
    float* cbuf = hbuf + 2 * (size_t)B_SZ * H_SZ;
    unsigned* bar = (unsigned*)(cbuf + 2 * (size_t)B_SZ * H_SZ);  // 64 u32
    const size_t bh = (size_t)B_SZ * H_SZ;
    size_t f32_bytes = ((size_t)B_SZ * T_SZ * H_SZ + 4 * bh) * 4 + 256;

    // ---- bf16 swizzled region ----
    const size_t x_el  = (size_t)B_SZ * T_SZ * H_SZ;
    const size_t w_el  = (size_t)T_SZ * 4 * H_SZ * H_SZ;
    const size_t we_el = (size_t)H_SZ * H_SZ;
    size_t bf_off = (f32_bytes + 63) & ~(size_t)63;
    __bf16* xb     = (__bf16*)((char*)d_ws + bf_off);
    __bf16* Wihb   = xb + x_el;
    __bf16* Whhb   = Wihb + w_el;
    __bf16* Wexitb = Whhb + w_el;
    size_t bf_end = bf_off + (x_el + 2 * w_el + we_el) * sizeof(__bf16);

    // ---- gates_pre region (tier A) ----
    size_t gp_off = (bf_end + 63) & ~(size_t)63;
    float* gp = (float*)((char*)d_ws + gp_off);
    size_t gp_end = gp_off + (size_t)T_SZ * B_SZ * 4 * H_SZ * sizeof(float);

    const bool pre     = (ws_size >= bf_end);
    const bool persist = (ws_size >= gp_end);

    if (pre) {
        long long tx = (long long)(x_el / 16);
        long long tw = (long long)(w_el / 16);
        long long te = (long long)(we_el / 16);
        swizzle_bf16_kernel<<<(unsigned)((tx + 255) / 256), 256, 0, stream>>>(x, xb, T_SZ * H_SZ, tx);
        swizzle_bf16_kernel<<<(unsigned)((tw + 255) / 256), 256, 0, stream>>>(W_ih, Wihb, H_SZ, tw);
        swizzle_bf16_kernel<<<(unsigned)((tw + 255) / 256), 256, 0, stream>>>(W_hh, Whhb, H_SZ, tw);
        swizzle_bf16_kernel<<<(unsigned)((te + 255) / 256), 256, 0, stream>>>(W_exit, Wexitb, H_SZ, te);
    }

    if (persist) {
        zero_bar_kernel<<<1, 64, 0, stream>>>(bar);
        gates_pre_kernel<<<dim3(64, T_SZ), 256, 0, stream>>>(xb, Wihb, b_ih, b_hh, gp);
        lstm_recurrent_kernel<<<NBLK_REC, 256, 0, stream>>>(Whhb, gp, hbuf, cbuf, hs, bar);
    } else {
        for (int t = 0; t < T_SZ; ++t) {
            const float* bih_t = b_ih + (size_t)t * 4 * H_SZ;
            const float* bhh_t = b_hh + (size_t)t * 4 * H_SZ;
            float* h_in  = hbuf + ((t + 1) & 1) * bh;
            float* h_out = hbuf + (t & 1) * bh;
            float* c_in  = cbuf + ((t + 1) & 1) * bh;
            float* c_out = cbuf + (t & 1) * bh;
            if (pre) {
                lstm_step_kernel<true><<<64, 256, 0, stream>>>(
                    x, xb, W_ih, Wihb, W_hh, Whhb, bih_t, bhh_t,
                    h_in, c_in, h_out, c_out, hs, t, (t == 0) ? 1 : 0);
            } else {
                lstm_step_kernel<false><<<64, 256, 0, stream>>>(
                    x, xb, W_ih, Wihb, W_hh, Whhb, bih_t, bhh_t,
                    h_in, c_in, h_out, c_out, hs, t, (t == 0) ? 1 : 0);
            }
        }
    }

    if (pre) exit_gemm_kernel<true><<<416, 256, 0, stream>>>(hs, W_exit, Wexitb, b_exit, out);
    else     exit_gemm_kernel<false><<<416, 256, 0, stream>>>(hs, W_exit, Wexitb, b_exit, out);
}
